// FixedNTLBGCore_32882269618912
// MI455X (gfx1250) — matrix operand order repaired and checked by reading, not compiled, not run on hardware
//
#include <hip/hip_runtime.h>
#include <math.h>

typedef __attribute__((ext_vector_type(8)))  float          v8f;
typedef __attribute__((ext_vector_type(16))) __bf16         v16bf;
typedef __attribute__((ext_vector_type(16))) unsigned short v16us;
typedef __attribute__((ext_vector_type(4)))  unsigned int   v4u;
typedef __attribute__((ext_vector_type(8)))  int            v8i;
typedef __attribute__((ext_vector_type(4)))  int            v4i;

#define Bq   128
#define Tt   1000
#define DV   256
#define KSEL 6
#define NROW (Bq*Tt)          // 128000 rows
// d_out float offsets: rep, idx, dist, mu_q, sigma, x
#define REP_OFF   0
#define IDX_OFF   196608
#define DIST_OFF  197376
#define MU_OFF    325376
#define SIG_OFF   358144
#define X_OFF     390912
// workspace byte offsets
#define WS_QPART  0            // 128*256 f32
#define WS_WT     131072       // 256*256 bf16 (ushort)
#define WS_REL    262144       // 128000 f32
#define WS_RSIG   774144       // 128*256 f32: 1/(sigma+1e-8)

#define WTPAD 264              // padded LDS row stride (ushorts): 512B + 16B pad

__device__ __forceinline__ unsigned short f2bf(float f) {
    unsigned u = __float_as_uint(f);
    u += 0x7FFFu + ((u >> 16) & 1u);
    return (unsigned short)(u >> 16);
}
__device__ __forceinline__ float gelu_exact(float x) {
    return 0.5f * x * (1.0f + erff(x * 0.70710678118654752f));
}
__device__ __forceinline__ float fast_rcp(float x) {
#if __has_builtin(__builtin_amdgcn_rcpf)
    return __builtin_amdgcn_rcpf(x);   // single v_rcp_f32, ~1 ulp
#else
    return 1.0f / x;
#endif
}
// branch-free gelu (Abramowitz-Stegun 7.1.26 erf, |err| < 1.5e-7) for the
// selection-score epilogue only (already bf16-quantized upstream).
__device__ __forceinline__ float gelu_fast(float x) {
    float z  = x * 0.70710678118654752f;
    float az = fabsf(z);
    float t  = fast_rcp(1.0f + 0.3275911f * az);
    float p  = t * (0.254829592f + t * (-0.284496736f +
               t * (1.421413741f + t * (-1.453152027f + t * 1.061405429f))));
    float er = 1.0f - p * __expf(-az * az);
    er = copysignf(er, z);
    return 0.5f * x * (1.0f + er);
}

// ---------------- prep: Wt[n*256+k] = bf16(sel_w1[k*256+n]) --------------
__global__ __launch_bounds__(256) void prep_wt(const float* __restrict__ sel_w1,
                                               unsigned short* __restrict__ Wt) {
    int e = blockIdx.x * 256 + threadIdx.x;   // 65536 total
    int n = e >> 8, k = e & 255;
    Wt[n * 256 + k] = f2bf(sel_w1[k * 256 + n]);
}

// ---------------- per-batch query MLPs: mu_q, sigma, qpart ---------------
__global__ __launch_bounds__(256) void qmlp(const float* __restrict__ q,
    const float* __restrict__ mu_w1, const float* __restrict__ mu_b1,
    const float* __restrict__ mu_w2, const float* __restrict__ mu_b2,
    const float* __restrict__ lng,   const float* __restrict__ lnb,
    const float* __restrict__ sg_w1, const float* __restrict__ sg_b1,
    const float* __restrict__ sg_w2, const float* __restrict__ sg_b2,
    const float* __restrict__ sel_w1,const float* __restrict__ sel_b1,
    float* __restrict__ out, float* __restrict__ qpart,
    float* __restrict__ rsig) {
    __shared__ float qs[256], buf[512], red[256];
    const int b = blockIdx.x, tid = threadIdx.x;
    qs[tid] = q[b * 256 + tid];
    __syncthreads();
    for (int jj = tid; jj < 512; jj += 256) {
        float s = mu_b1[jj];
        for (int k = 0; k < 256; ++k) s += qs[k] * mu_w1[k * 512 + jj];
        buf[jj] = gelu_exact(s);
    }
    __syncthreads();
    float s = mu_b2[tid];
    for (int k = 0; k < 512; ++k) s += buf[k] * mu_w2[k * 256 + tid];
    red[tid] = s; __syncthreads();
    for (int o = 128; o > 0; o >>= 1) { if (tid < o) red[tid] += red[tid + o]; __syncthreads(); }
    float mean = red[0] * (1.0f / 256.0f); __syncthreads();
    float dv = s - mean;
    red[tid] = dv * dv; __syncthreads();
    for (int o = 128; o > 0; o >>= 1) { if (tid < o) red[tid] += red[tid + o]; __syncthreads(); }
    float var = red[0] * (1.0f / 256.0f); __syncthreads();
    out[MU_OFF + b * 256 + tid] = dv * rsqrtf(var + 1e-5f) * lng[tid] + lnb[tid];
    float s1 = sg_b1[tid];
    for (int k = 0; k < 256; ++k) s1 += qs[k] * sg_w1[k * 256 + tid];
    buf[tid] = gelu_exact(s1);
    __syncthreads();
    float s2 = sg_b2[tid];
    for (int k = 0; k < 256; ++k) s2 += buf[k] * sg_w2[k * 256 + tid];
    float sig = 2.0f / (1.0f + __expf(-s2)) + 0.1f;
    out[SIG_OFF + b * 256 + tid] = sig;
    rsig[b * 256 + tid] = 1.0f / (sig + 1e-8f);   // exact, hoisted off hot path
    float s3 = sel_b1[tid];
    for (int k = 0; k < 256; ++k) s3 += qs[k] * sel_w1[(256 + k) * 256 + tid];
    qpart[b * 256 + tid] = s3;
}

// -------- main fused kernel: x, dist, and WMMA selection GEMM -> rel ------
__global__ __launch_bounds__(256) void main_fused(
    const float* __restrict__ video, const float* __restrict__ pos,
    const unsigned short* __restrict__ Wt, const float* __restrict__ qpart,
    const float* __restrict__ rsig,
    const float* __restrict__ sel_w2, const float* __restrict__ sel_b2,
    float* __restrict__ out, float* __restrict__ rel) {
    __shared__ float dist_s[128], rel_s[128];
    __shared__ unsigned short wts[64][WTPAD];   // one 64-col chunk of Wt, padded
    const int tid = threadIdx.x;
    const int g0base = blockIdx.x * 128;
    if (tid < 128) { dist_s[tid] = 0.0f; rel_s[tid] = 0.0f; }
    __syncthreads();

    // ---- phase A: x = video + pos (write out), fused dist reduction ----
    {
        const int m = tid >> 1;
        const int g = g0base + m;
        const int t = g % Tt;
        const int b = g / Tt;
        const int k0 = (tid & 1) * 128;
        const float4* vv = (const float4*)(video + (size_t)g * 256 + k0);
        const float4* pp = (const float4*)(pos + t * 256 + k0);
        const float4* mm = (const float4*)(out + MU_OFF + b * 256 + k0);
        const float4* rr = (const float4*)(rsig + b * 256 + k0);
        float4* xo = (float4*)(out + X_OFF + (size_t)g * 256 + k0);
        float acc = 0.0f;
        #pragma unroll 4
        for (int i = 0; i < 32; ++i) {
            float4 a = vv[i], p = pp[i], mu = mm[i], rs = rr[i];
            float4 x;
            x.x = a.x + p.x; x.y = a.y + p.y; x.z = a.z + p.z; x.w = a.w + p.w;
            xo[i] = x;
            float cx = x.x - mu.x, cy = x.y - mu.y, cz = x.z - mu.z, cw = x.w - mu.w;
            acc += cx * cx * rs.x + cy * cy * rs.y + cz * cz * rs.z + cw * cw * rs.w;
        }
        atomicAdd(&dist_s[m], acc);
    }
    __syncthreads();
    if (tid < 128) out[DIST_OFF + g0base + tid] = fmaxf(dist_s[tid], 1e-8f);

    // ---- phase B: h = x @ W1x via WMMA bf16, epilogue gelu . w2 ----
    const int lane = tid & 31;
    const int wave = tid >> 5;            // 8 waves -> 8 M-tiles of 16 rows
    const int nloc = lane & 15;
    const int half = lane >> 4;
    const int g0 = g0base + wave * 16;

    // A fragments (ISA 16-bit A layout): row = lane&15; per k-step two
    // contiguous 8-wide K runs at kbase+8*half and kbase+16+8*half.
    const int arow = g0 + nloc;
    const int t_a = arow % Tt;
    const float* vrow = video + (size_t)arow * 256;
    const float* prow = pos + t_a * 256;
    union AU { v16us u; v16bf b; };
    v16bf afrag[8];
    #pragma unroll
    for (int kk = 0; kk < 8; ++kk) {
        const int k0 = kk * 32 + 8 * half;
        const int k1 = k0 + 16;
        float4 va = *(const float4*)(vrow + k0), vb = *(const float4*)(vrow + k0 + 4);
        float4 pa = *(const float4*)(prow + k0), pb = *(const float4*)(prow + k0 + 4);
        float4 vc = *(const float4*)(vrow + k1), vd = *(const float4*)(vrow + k1 + 4);
        float4 pc = *(const float4*)(prow + k1), pd = *(const float4*)(prow + k1 + 4);
        AU A;
        A.u[0]  = f2bf(va.x + pa.x); A.u[1]  = f2bf(va.y + pa.y);
        A.u[2]  = f2bf(va.z + pa.z); A.u[3]  = f2bf(va.w + pa.w);
        A.u[4]  = f2bf(vb.x + pb.x); A.u[5]  = f2bf(vb.y + pb.y);
        A.u[6]  = f2bf(vb.z + pb.z); A.u[7]  = f2bf(vb.w + pb.w);
        A.u[8]  = f2bf(vc.x + pc.x); A.u[9]  = f2bf(vc.y + pc.y);
        A.u[10] = f2bf(vc.z + pc.z); A.u[11] = f2bf(vc.w + pc.w);
        A.u[12] = f2bf(vd.x + pd.x); A.u[13] = f2bf(vd.y + pd.y);
        A.u[14] = f2bf(vd.z + pd.z); A.u[15] = f2bf(vd.w + pd.w);
        afrag[kk] = A.b;
    }

    float racc[8] = {0, 0, 0, 0, 0, 0, 0, 0};
    for (int c = 0; c < 4; ++c) {           // 4 chunks of 64 N-columns
        const int n0 = c * 64;
        __syncthreads();                    // previous chunk fully consumed
#if __has_builtin(__builtin_amdgcn_tensor_load_to_lds) && __has_builtin(__builtin_amdgcn_s_wait_tensorcnt)
        // TDM: DMA Wt[n0:n0+64, 0:256] (bf16) global->LDS with 16B row pad.
        // Wave-uniform branch so exactly one wave issues (TDM ignores EXEC).
        if ((__builtin_amdgcn_readfirstlane(tid) >> 5) == 0) {
            unsigned ldsb = (unsigned)(unsigned long long)
                (__attribute__((address_space(3))) unsigned short*)&wts[0][0];
            unsigned long long ga = (unsigned long long)(Wt + (size_t)n0 * 256);
            v4u g0d = { 1u,                                  // count=1, load
                        ldsb,                                // lds_addr
                        (unsigned)ga,                        // global_addr[31:0]
                        ((unsigned)(ga >> 32) & 0x01FFFFFFu) // global_addr[56:32]
                        | 0x80000000u };                     // type=2
            int w0 = (1 << 16)        // data_size = 2B
                   | (1 << 20)        // pad_enable
                   | (6 << 22)        // pad_interval = 128 DWORDs (512B row)
                   | (3 << 25);       // pad_amount   = 4 DWORDs (16B)
            v8i g1d = { w0,
                        (int)(256u << 16),          // tensor_dim0 lo16 (=256)
                        (int)(64u << 16),           // dim0 hi16=0 | tensor_dim1 lo16=64
                        (int)(256u << 16),          // dim1 hi16=0 | tile_dim0=256
                        64,                         // tile_dim1=64, tile_dim2=0
                        256,                        // tensor_dim0_stride lo32
                        0, 0 };                     // stride hi | dim1_stride
            v4i g2d = { 0, 0, 0, 0 };
            v4i g3d = { 0, 0, 0, 0 };
            v8i g4d = { 0, 0, 0, 0, 0, 0, 0, 0 };   // extra group (6-arg form)
            __builtin_amdgcn_tensor_load_to_lds(g0d, g1d, g2d, g3d, g4d, 0);
            __builtin_amdgcn_s_wait_tensorcnt(0);
        }
#else
        for (int e = tid; e < 64 * 32; e += 256) {  // sync fallback staging
            int row = e >> 5, seg = e & 31;
            *(uint4*)(&wts[row][seg * 8]) =
                *(const uint4*)(Wt + (size_t)(n0 + row) * 256 + seg * 8);
        }
#endif
        __syncthreads();

        for (int nt = 0; nt < 4; ++nt) {
            const int n = n0 + nt * 16 + nloc;      // B column for this lane
            const unsigned short* brow = &wts[nt * 16 + nloc][8 * half];
            v8f acc = {};
            #pragma unroll
            for (int kk = 0; kk < 8; ++kk) {
                // B layout: lane = column n, element e -> K = 8*half + e (e<8), 16 + 8*half + (e-8) (e>=8)
                union BU { uint4 q[2]; v16bf b; } Bf;
                const uint4* bp = (const uint4*)(brow + kk * 32);
                Bf.q[0] = bp[0]; Bf.q[1] = bp[2];
                acc = __builtin_amdgcn_wmma_f32_16x16x32_bf16(
                    false, afrag[kk], false, Bf.b, (short)0, acc, false, false);
            }
            const float w2v = sel_w2[n];
            #pragma unroll
            for (int r = 0; r < 8; ++r) {           // D: VGPR r, M = r + 8*half
                const int g = g0 + r + 8 * half;
                const int bb = g / Tt;
                float h = acc[r] + qpart[bb * 256 + n];
                racc[r] += gelu_fast(h) * w2v;
            }
        }
    }
    #pragma unroll
    for (int r = 0; r < 8; ++r)
        atomicAdd(&rel_s[wave * 16 + r + 8 * half], racc[r]);
    __syncthreads();
    if (tid < 128) rel[g0base + tid] = rel_s[tid] + sel_b2[0];
}

// -------- per-batch: median, comb, top-12, diversify, gather ------------
__global__ __launch_bounds__(256) void select_k(const float* __restrict__ rel,
                                                float* __restrict__ out) {
    __shared__ float dl[Tt], cb[Tt], rv[256], medS;
    __shared__ int ri[256], cand[2 * KSEL], selS[KSEL];
    const int b = blockIdx.x, tid = threadIdx.x;
    for (int t = tid; t < Tt; t += 256) dl[t] = out[DIST_OFF + b * Tt + t];
    __syncthreads();
    for (int t = tid; t < Tt; t += 256) {
        float v = dl[t]; int cl = 0, ce = 0;
        for (int s = 0; s < Tt; ++s) { cl += dl[s] < v; ce += dl[s] == v; }
        if (cl <= 499 && cl + ce > 499) medS = v;
    }
    __syncthreads();
    float med = medS;
    for (int t = tid; t < Tt; t += 256)
        cb[t] = 0.5f * rel[b * Tt + t] - fabsf(dl[t] - med);
    __syncthreads();
    for (int it = 0; it < 2 * KSEL; ++it) {
        float bv = -3.402823466e38f; int bi = 1 << 30;
        for (int t = tid; t < Tt; t += 256) {
            float v = cb[t];
            if (v > bv || (v == bv && t < bi)) { bv = v; bi = t; }
        }
        rv[tid] = bv; ri[tid] = bi; __syncthreads();
        for (int o = 128; o > 0; o >>= 1) {
            if (tid < o) {
                if (rv[tid + o] > rv[tid] ||
                    (rv[tid + o] == rv[tid] && ri[tid + o] < ri[tid])) {
                    rv[tid] = rv[tid + o]; ri[tid] = ri[tid + o];
                }
            }
            __syncthreads();
        }
        if (tid == 0) { cand[it] = ri[0]; cb[ri[0]] = -3.402823466e38f; }
        __syncthreads();
    }
    if (tid == 0) {
        int cs[2 * KSEL];
        for (int i = 0; i < 2 * KSEL; ++i) cs[i] = cand[i];
        for (int i = 1; i < 2 * KSEL; ++i) {
            int key = cs[i], j = i - 1;
            while (j >= 0 && cs[j] > key) { cs[j + 1] = cs[j]; --j; }
            cs[j + 1] = key;
        }
        int md[2 * KSEL]; bool rem[2 * KSEL];
        const int s0 = cs[0];
        for (int i = 0; i < 2 * KSEL; ++i) { md[i] = abs(cs[i] - s0); rem[i] = (i != 0); }
        selS[0] = s0;
        for (int step = 1; step < KSEL; ++step) {
            int best = -2, bi2 = 0;
            for (int i = 0; i < 2 * KSEL; ++i) {
                int sc = rem[i] ? md[i] : -1;
                if (sc > best) { best = sc; bi2 = i; }
            }
            int s = cs[bi2];
            rem[bi2] = false;
            for (int i = 0; i < 2 * KSEL; ++i) md[i] = min(md[i], abs(cs[i] - s));
            selS[step] = s;
        }
    }
    __syncthreads();
    for (int kk = 0; kk < KSEL; ++kk) {
        int t = min(max(selS[kk], 0), Tt - 1);
        out[REP_OFF + (b * KSEL + kk) * 256 + tid] =
            out[X_OFF + ((size_t)b * Tt + t) * 256 + tid];
    }
    if (tid < KSEL) out[IDX_OFF + b * KSEL + tid] = (float)selS[tid];
}

extern "C" void kernel_launch(void* const* d_in, const int* in_sizes, int n_in,
                              void* d_out, int out_size, void* d_ws, size_t ws_size,
                              hipStream_t stream) {
    const float* video  = (const float*)d_in[0];
    const float* query  = (const float*)d_in[1];
    const float* pos    = (const float*)d_in[2];
    const float* mu_w1  = (const float*)d_in[3];
    const float* mu_b1  = (const float*)d_in[4];
    const float* mu_w2  = (const float*)d_in[5];
    const float* mu_b2  = (const float*)d_in[6];
    const float* lng    = (const float*)d_in[7];
    const float* lnb    = (const float*)d_in[8];
    const float* sg_w1  = (const float*)d_in[9];
    const float* sg_b1  = (const float*)d_in[10];
    const float* sg_w2  = (const float*)d_in[11];
    const float* sg_b2  = (const float*)d_in[12];
    const float* sel_w1 = (const float*)d_in[13];
    const float* sel_b1 = (const float*)d_in[14];
    const float* sel_w2 = (const float*)d_in[15];
    const float* sel_b2 = (const float*)d_in[16];
    float* out = (float*)d_out;
    char* ws = (char*)d_ws;
    float* qpart = (float*)(ws + WS_QPART);
    unsigned short* Wt = (unsigned short*)(ws + WS_WT);
    float* rel = (float*)(ws + WS_REL);
    float* rsig = (float*)(ws + WS_RSIG);

    prep_wt<<<256, 256, 0, stream>>>(sel_w1, Wt);
    qmlp<<<Bq, 256, 0, stream>>>(query, mu_w1, mu_b1, mu_w2, mu_b2, lng, lnb,
                                 sg_w1, sg_b1, sg_w2, sg_b2, sel_w1, sel_b1,
                                 out, qpart, rsig);
    main_fused<<<NROW / 128, 256, 0, stream>>>(video, pos, Wt, qpart, rsig,
                                               sel_w2, sel_b2, out, rel);
    select_k<<<Bq, 256, 0, stream>>>(rel, out);
}
